// StateNetV5_79860621902552
// MI455X (gfx1250) — compile-verified, matrix operand order checked by reading, not run on hardware
//
#include <hip/hip_runtime.h>

typedef __attribute__((ext_vector_type(16))) __bf16 v16bf;
typedef __attribute__((ext_vector_type(8)))  float  v8f;

#define NB      (1u << 20)
#define NTILES  (NB / 16)
#define SPITCH  68   // 16B-aligned rows; 68*L mod 64 distinct for L=0..15

__device__ __forceinline__ void lds_fence() {
  // hardware keeps same-wave DS ops in order; this stops the compiler from
  // reordering cross-lane LDS traffic and forces the dscnt wait explicitly.
  asm volatile("s_wait_dscnt 0x0" ::: "memory");
}

__device__ __forceinline__ v8f wmma_bf16(v16bf a, v16bf b, v8f c) {
  return __builtin_amdgcn_wmma_f32_16x16x32_bf16(
      /*neg_a=*/false, a, /*neg_b=*/false, b,
      /*c_mod=*/(short)0, c, /*reuse_a=*/false, /*reuse_b=*/false);
}

__global__ __launch_bounds__(256)
void statenet_kernel(const float* __restrict__ state,
                     const float* __restrict__ W_enc1, const float* __restrict__ b_enc1,
                     const float* __restrict__ ln_g,   const float* __restrict__ ln_b,
                     const float* __restrict__ W_enc2, const float* __restrict__ b_enc2,
                     const float* __restrict__ W_enc3, const float* __restrict__ b_enc3,
                     const float* __restrict__ W_dec1, const float* __restrict__ b_dec1,
                     const float* __restrict__ W_dec2, const float* __restrict__ b_dec2,
                     const float* __restrict__ W_m,    const float* __restrict__ b_m,
                     const float* __restrict__ W_h,    const float* __restrict__ b_h,
                     const float* __restrict__ W_c,    const float* __restrict__ b_c,
                     const float* __restrict__ s_m,    const float* __restrict__ s_h,
                     const float* __restrict__ s_c,
                     float* __restrict__ out)
{
  // 22 pre-swizzled WMMA B tiles (bf16, per-lane layout), LN params, per-wave scratch
  __shared__ __align__(32) __bf16 lB[22][32][16];
  __shared__ float lG[64], lBt[64];
  __shared__ float lS[8][16 * SPITCH];

  const int tid = threadIdx.x;

  // ---------------- one-time weight pre-swizzle into LDS (WMMA B layout) ----------
  // B (32x16) per-lane element map: lane<16 holds N=lane, K in {0..7,16..23};
  // lane>=16 holds N=lane-16, K in {8..15,24..31}; element e -> local K below.
  for (int i = tid; i < 22 * 32; i += 256) {
    int t = i >> 5, ln = i & 31;
    int col = ln & 15, hf = ln >> 4;
    const float* W = nullptr; int ld = 0, ncols = 0, nb = 0, kb = 0, kv = 0;
    if (t >= 2) {
      if (t < 6)       { W = W_enc1; ld = 64; ncols = 64; nb = (t - 2) * 16;  kb = 0;              kv = 21; }
      else if (t < 14) { int j = t - 6; W = W_enc2; ld = 64; ncols = 64; nb = (j >> 1) * 16; kb = (j & 1) * 32; kv = 64; }
      else if (t < 16) { W = W_enc3; ld = 14; ncols = 14; nb = 0;             kb = (t - 14) * 32;  kv = 64; }
      else if (t < 20) { W = W_dec1; ld = 64; ncols = 64; nb = (t - 16) * 16; kb = 0;              kv = 14; }
      else             { W = W_dec2; ld = 8;  ncols = 8;  nb = 0;             kb = (t - 20) * 32;  kv = 64; }
    }
    #pragma unroll
    for (int e = 0; e < 16; ++e) {
      int kl = hf ? (e < 8 ? e + 8 : e + 16) : (e < 8 ? e : e + 8);
      float v = 0.f;
      if (t < 2) {                      // packed attention heads: [W_m | W_h | W_c], 23 cols
        int c = t * 16 + col;
        if (kl < 21 && c < 23) {
          if (c < 13)      v = W_m[kl * 13 + c];
          else if (c < 19) v = W_h[kl * 6 + (c - 13)];
          else             v = W_c[kl * 4 + (c - 19)];
        }
      } else {
        int c = nb + col, k = kb + kl;
        if (k < kv && c < ncols) v = W[k * ld + c];
      }
      lB[t][ln][e] = (__bf16)v;
    }
  }
  for (int i = tid; i < 64; i += 256) { lG[i] = ln_g[i]; lBt[i] = ln_b[i]; }
  __syncthreads();

  // ---------------- per-wave setup ----------------
  const int lane = tid & 31;
  const int wv   = tid >> 5;
  const int col  = lane & 15;   // N column (C/D layout) / M row (A layout)
  const int hf   = lane >> 4;
  float* Sw = lS[wv];

  float bE1[4], bE2[4], bD1[4];
  #pragma unroll
  for (int n = 0; n < 4; ++n) {
    bE1[n] = b_enc1[n * 16 + col];
    bE2[n] = b_enc2[n * 16 + col];
    bD1[n] = b_dec1[n * 16 + col];
  }
  const float bE3 = (col < 14) ? b_enc3[col] : 0.f;
  const float bD2 = (col < 8)  ? b_dec2[col] : 0.f;
  const float bA0 = (col < 13) ? b_m[col] : b_h[col - 13];
  const float bA1 = (col < 3) ? b_h[col + 3] : ((col < 7) ? b_c[col - 3] : 0.f);
  const float scM = 2.f / (1.f + __expf(-s_m[0]));
  const float scH = 2.f / (1.f + __expf(-s_h[0]));
  const float scC = 2.f / (1.f + __expf(-s_c[0]));

  float* o_corr = out;
  float* o_lat  = out + (size_t)NB * 8;
  float* o_met  = out + (size_t)NB * 22;
  float* o_hyp  = out + (size_t)NB * 35;
  float* o_cur  = out + (size_t)NB * 41;

  const int gw      = blockIdx.x * 8 + wv;
  const int gstride = gridDim.x * 8;
  const v8f z = {};

  for (int t = gw; t < NTILES; t += gstride) {
    const size_t m0 = (size_t)t * 16;
    const int m  = col;      // this lane's row in A layout
    const int mo = hf * 8;   // this lane's row base in C/D layout
    const int c0 = hf * 8;   // this lane's K-run base

    // ---- load + normalize state, place in WMMA A layout (K padded 21->32) ----
    float nsv[16];
    {
      const float* rowp = state + (m0 + m) * 21;
      if (hf == 0) {           // K = 0..7, 16..23
        #pragma unroll
        for (int j = 0; j < 8; ++j) nsv[j] = rowp[j];
        nsv[8] = rowp[16]; nsv[9] = rowp[17]; nsv[10] = rowp[18];
        nsv[11] = rowp[19]; nsv[12] = rowp[20];
        nsv[13] = 0.f; nsv[14] = 0.f; nsv[15] = 0.f;
        nsv[0] *= (1.f / 3.f); nsv[1] *= (1.f / 3.f);      // cols 0,1 /3
        nsv[9] *= 0.5f;                                     // col 17 /2
        nsv[10] *= 0.25f;                                   // col 18 /4
        nsv[11] = fminf(fmaxf(nsv[11], 0.f), 2.f) * 0.5f;   // col 19 clip/2
      } else {                 // K = 8..15, 24..31 (>=21 padded)
        #pragma unroll
        for (int j = 0; j < 8; ++j) nsv[j] = rowp[8 + j];
        #pragma unroll
        for (int j = 8; j < 16; ++j) nsv[j] = 0.f;
      }
    }
    v16bf aNS;
    #pragma unroll
    for (int e = 0; e < 16; ++e) aNS[e] = (__bf16)nsv[e];

    // ---- attention logits: [16,32] x [32,23] via two WMMAs ----
    v8f cA0 = wmma_bf16(aNS, *(const v16bf*)lB[0][lane], z);
    v8f cA1 = wmma_bf16(aNS, *(const v16bf*)lB[1][lane], z);
    #pragma unroll
    for (int r = 0; r < 8; ++r) {
      Sw[(mo + r) * SPITCH + col]      = cA0[r] + bA0;
      Sw[(mo + r) * SPITCH + 16 + col] = cA1[r] + bA1;
    }
    lds_fence();

    // ---- softmaxes: every lane reads its row (redundant across halves) ----
    float lg[23];
    {
      const float* srow = Sw + m * SPITCH;
      #pragma unroll
      for (int c = 0; c < 23; ++c) lg[c] = srow[c];
    }
    float met[13], hyv[6], cuv[4];
    {
      float mx = lg[0];
      #pragma unroll
      for (int c = 1; c < 13; ++c) mx = fmaxf(mx, lg[c]);
      float s = 0.f;
      #pragma unroll
      for (int c = 0; c < 13; ++c) { met[c] = __expf(lg[c] - mx); s += met[c]; }
      float inv = scM / s;
      #pragma unroll
      for (int c = 0; c < 13; ++c) met[c] *= inv;
    }
    {
      float mx = lg[13];
      #pragma unroll
      for (int c = 14; c < 19; ++c) mx = fmaxf(mx, lg[c]);
      float s = 0.f;
      #pragma unroll
      for (int c = 0; c < 6; ++c) { hyv[c] = __expf(lg[13 + c] - mx); s += hyv[c]; }
      float inv = scH / s;
      #pragma unroll
      for (int c = 0; c < 6; ++c) hyv[c] *= inv;
    }
    {
      float mx = lg[19];
      #pragma unroll
      for (int c = 20; c < 23; ++c) mx = fmaxf(mx, lg[c]);
      float s = 0.f;
      #pragma unroll
      for (int c = 0; c < 4; ++c) { cuv[c] = __expf(lg[19 + c] - mx); s += cuv[c]; }
      float inv = scC / s;
      #pragma unroll
      for (int c = 0; c < 4; ++c) cuv[c] *= inv;
    }
    if (hf == 0) {             // one writer per row
      const size_t r = m0 + m;
      #pragma unroll
      for (int c = 0; c < 13; ++c) o_met[r * 13 + c] = met[c];
      #pragma unroll
      for (int c = 0; c < 6; ++c)  o_hyp[r * 6 + c]  = hyv[c];
      #pragma unroll
      for (int c = 0; c < 4; ++c)  o_cur[r * 4 + c]  = cuv[c];
    }

    // ---- attended = ns * metric[EXPAND_IDX[K]]  (per-lane K subset) ----
    float att[16];
    if (hf == 0) {   // K: 0..7 -> {0,1,2,3,4,4,5,5}; 16..20 -> {9,10,10,11,12}
      const int em[16] = {0,1,2,3,4,4,5,5, 9,10,10,11,12, -1,-1,-1};
      #pragma unroll
      for (int e = 0; e < 16; ++e) att[e] = (em[e] >= 0) ? nsv[e] * met[em[e]] : 0.f;
    } else {         // K: 8..15 -> {5,6,6,6,7,7,8,9}
      const int em[16] = {5,6,6,6,7,7,8,9, -1,-1,-1,-1,-1,-1,-1,-1};
      #pragma unroll
      for (int e = 0; e < 16; ++e) att[e] = (em[e] >= 0) ? nsv[e] * met[em[e]] : 0.f;
    }
    v16bf aATT;
    #pragma unroll
    for (int e = 0; e < 16; ++e) aATT[e] = (__bf16)att[e];

    lds_fence();   // logits reads done before enc1 overwrites scratch

    // ---- enc1: [16,32]x[32,64] -> LDS (pre-LN) ----
    #pragma unroll
    for (int n = 0; n < 4; ++n) {
      v8f c = wmma_bf16(aATT, *(const v16bf*)lB[2 + n][lane], z);
      #pragma unroll
      for (int r = 0; r < 8; ++r) Sw[(mo + r) * SPITCH + n * 16 + col] = c[r] + bE1[n];
    }
    lds_fence();

    // ---- layernorm(64) + tanh, rebuild A tiles (K=64 -> a0,a1) ----
    float hv[32];
    #pragma unroll
    for (int run = 0; run < 4; ++run)
      #pragma unroll
      for (int j = 0; j < 8; ++j)
        hv[run * 8 + j] = Sw[m * SPITCH + c0 + run * 16 + j];
    {
      float s1 = 0.f;
      #pragma unroll
      for (int j = 0; j < 32; ++j) s1 += hv[j];
      s1 += __shfl_xor(s1, 16, 32);               // combine lane pair (rows split over 2 lanes)
      const float mu = s1 * (1.f / 64.f);
      float s2 = 0.f;
      #pragma unroll
      for (int j = 0; j < 32; ++j) { float d = hv[j] - mu; s2 += d * d; }
      s2 += __shfl_xor(s2, 16, 32);
      const float rstd = rsqrtf(s2 * (1.f / 64.f) + 1e-5f);
      #pragma unroll
      for (int run = 0; run < 4; ++run)
        #pragma unroll
        for (int j = 0; j < 8; ++j) {
          int cix = c0 + run * 16 + j;
          int v = run * 8 + j;
          hv[v] = tanhf((hv[v] - mu) * rstd * lG[cix] + lBt[cix]);
        }
    }
    v16bf a0, a1;
    #pragma unroll
    for (int e = 0; e < 16; ++e) { a0[e] = (__bf16)hv[e]; a1[e] = (__bf16)hv[16 + e]; }

    lds_fence();

    // ---- enc2: [16,64]x[64,64], relu -> LDS ----
    #pragma unroll
    for (int n = 0; n < 4; ++n) {
      v8f c = wmma_bf16(a0, *(const v16bf*)lB[6 + 2 * n][lane], z);
      c     = wmma_bf16(a1, *(const v16bf*)lB[7 + 2 * n][lane], c);
      #pragma unroll
      for (int r = 0; r < 8; ++r)
        Sw[(mo + r) * SPITCH + n * 16 + col] = fmaxf(c[r] + bE2[n], 0.f);
    }
    lds_fence();

    #pragma unroll
    for (int run = 0; run < 4; ++run)
      #pragma unroll
      for (int j = 0; j < 8; ++j)
        hv[run * 8 + j] = Sw[m * SPITCH + c0 + run * 16 + j];
    #pragma unroll
    for (int e = 0; e < 16; ++e) { a0[e] = (__bf16)hv[e]; a1[e] = (__bf16)hv[16 + e]; }

    // ---- enc3: latent = [16,64]x[64,14] ----
    v8f cl = wmma_bf16(a0, *(const v16bf*)lB[14][lane], z);
    cl     = wmma_bf16(a1, *(const v16bf*)lB[15][lane], cl);
    lds_fence();   // enc2 reads done before latent overwrites cols 0..15
    #pragma unroll
    for (int r = 0; r < 8; ++r) {
      float v = cl[r] + bE3;                       // cols 14,15: zero-padded B + 0 bias -> 0
      Sw[(mo + r) * SPITCH + col] = v;
      if (col < 14) o_lat[(m0 + mo + r) * 14 + col] = v;
    }
    lds_fence();

    // ---- dec1 A tile: K=14 padded to 32 ----
    float lv[16];
    #pragma unroll
    for (int j = 0; j < 8; ++j) lv[j] = Sw[m * SPITCH + c0 + j];
    #pragma unroll
    for (int j = 8; j < 16; ++j) lv[j] = 0.f;
    v16bf aL;
    #pragma unroll
    for (int e = 0; e < 16; ++e) aL[e] = (__bf16)lv[e];
    lds_fence();

    // ---- dec1: [16,32]x[32,64], relu -> LDS ----
    #pragma unroll
    for (int n = 0; n < 4; ++n) {
      v8f c = wmma_bf16(aL, *(const v16bf*)lB[16 + n][lane], z);
      #pragma unroll
      for (int r = 0; r < 8; ++r)
        Sw[(mo + r) * SPITCH + n * 16 + col] = fmaxf(c[r] + bD1[n], 0.f);
    }
    lds_fence();

    #pragma unroll
    for (int run = 0; run < 4; ++run)
      #pragma unroll
      for (int j = 0; j < 8; ++j)
        hv[run * 8 + j] = Sw[m * SPITCH + c0 + run * 16 + j];
    #pragma unroll
    for (int e = 0; e < 16; ++e) { a0[e] = (__bf16)hv[e]; a1[e] = (__bf16)hv[16 + e]; }

    // ---- dec2: corrections = tanh([16,64]x[64,8]) ----
    v8f cc = wmma_bf16(a0, *(const v16bf*)lB[20][lane], z);
    cc     = wmma_bf16(a1, *(const v16bf*)lB[21][lane], cc);
    #pragma unroll
    for (int r = 0; r < 8; ++r)
      if (col < 8) o_corr[(m0 + mo + r) * 8 + col] = tanhf(cc[r] + bD2);

    lds_fence();   // dec1 reads done before next iteration's logits stores
  }
}

extern "C" void kernel_launch(void* const* d_in, const int* in_sizes, int n_in,
                              void* d_out, int out_size, void* d_ws, size_t ws_size,
                              hipStream_t stream) {
  (void)in_sizes; (void)n_in; (void)out_size; (void)d_ws; (void)ws_size;
  const float* p[22];
  for (int i = 0; i < 22; ++i) p[i] = (const float*)d_in[i];
  dim3 grid(1024), block(256);
  hipLaunchKernelGGL(statenet_kernel, grid, block, 0, stream,
                     p[0], p[1], p[2], p[3], p[4], p[5], p[6], p[7], p[8], p[9],
                     p[10], p[11], p[12], p[13], p[14], p[15], p[16], p[17],
                     p[18], p[19], p[20], p[21], (float*)d_out);
}
